// GRUD_cell_39230231281830
// MI455X (gfx1250) — compile-verified, matrix operand order checked by reading, not run on hardware
//
#include <hip/hip_runtime.h>

// ---------------- problem constants ----------------
#define Bk 256   // batch
#define Ik 128   // input features
#define Tk 256   // time steps
#define Hk 512   // hidden
#define Ok 2     // output
#define ACW 1280 // A_cat width: [x(128) | hdec(512) | m(128) | rh(512)]

#define KS   128 // K-slab staged per barrier (4 WMMA k-steps)
#define BSTR 132 // LDS row stride for B tile (keeps uint2 stores 8B aligned)

typedef __attribute__((ext_vector_type(16))) __bf16 bf16x16;
typedef __attribute__((ext_vector_type(8)))  float  floatx8;

union FragAB { bf16x16 v; unsigned int u[8]; };

__device__ __forceinline__ unsigned short f2bf(float f) {
  unsigned int u = __float_as_uint(f);
  u += 0x7FFFu + ((u >> 16) & 1u);          // round-to-nearest-even
  return (unsigned short)(u >> 16);
}

// ---------------- pipelined WMMA GEMM core ----------------
// Block: 128 threads = 4 waves (wave32). Block tile: 16 rows x 64 cols.
// Each wave owns one 16x16 f32 accumulator (v_wmma_f32_16x16x32_bf16).
// K is processed in 128-wide slabs: register-prefetch next slab's global
// data, run 4 WMMAs out of the current LDS buffer, then commit the prefetch
// to the alternate LDS buffer (double buffered), one barrier per slab.
// aload(r, k4) returns 4 packed bf16 of A[m0+r][k4..k4+3].
// BT is COLUMN-major bf16: BT[n*KTOT + k].
template <int KTOT, class ALoad>
__device__ __forceinline__ floatx8
gemm_pipe(ALoad aload, const unsigned short* __restrict__ BT, int n0,
          unsigned short* lds_a, unsigned short* lds_b)
{
  const int tid  = threadIdx.x;
  const int w    = tid >> 5;
  const int lane = tid & 31;
  const int hi   = lane >> 4;
  const int lq   = lane & 15;
  constexpr int NSLAB = KTOT / KS;
  floatx8 acc = {0.f, 0.f, 0.f, 0.f, 0.f, 0.f, 0.f, 0.f};

  uint2 ra[4];   // A prefetch: 16 rows x 128 k / 128 thr = 4 x uint2
  uint2 rb[16];  // B prefetch: 64 n x 128 k / 128 thr = 16 x uint2

  // ---- prologue: slab 0 straight into LDS buffer 0 ----
#pragma unroll
  for (int it = 0; it < 4; ++it) {
    int idx = tid + 128 * it, r = idx >> 5, k = (idx & 31) * 4;
    ra[it] = aload(r, k);
  }
#pragma unroll
  for (int it = 0; it < 16; ++it) {
    int idx = tid + 128 * it, n = idx >> 5, k = (idx & 31) * 4;
    rb[it] = *(const uint2*)&BT[(size_t)(n0 + n) * KTOT + k];
  }
#pragma unroll
  for (int it = 0; it < 4; ++it) {
    int idx = tid + 128 * it, r = idx >> 5, k = (idx & 31) * 4;
    *(uint2*)&lds_a[r * KS + k] = ra[it];
  }
#pragma unroll
  for (int it = 0; it < 16; ++it) {
    int idx = tid + 128 * it, n = idx >> 5, k = (idx & 31) * 4;
    *(uint2*)&lds_b[n * BSTR + k] = rb[it];
  }
  __syncthreads();

#pragma unroll
  for (int s = 0; s < NSLAB; ++s) {
    unsigned short* ca = lds_a + (s & 1) * (16 * KS);
    unsigned short* cb = lds_b + (s & 1) * (64 * BSTR);

    // issue next slab's global loads (no wait needed until the LDS commit)
    if (s + 1 < NSLAB) {
      const int kt = (s + 1) * KS;
#pragma unroll
      for (int it = 0; it < 4; ++it) {
        int idx = tid + 128 * it, r = idx >> 5, k = (idx & 31) * 4;
        ra[it] = aload(r, kt + k);
      }
#pragma unroll
      for (int it = 0; it < 16; ++it) {
        int idx = tid + 128 * it, n = idx >> 5, k = (idx & 31) * 4;
        rb[it] = *(const uint2*)&BT[(size_t)(n0 + n) * KTOT + kt + k];
      }
    }

    // 4 WMMAs out of the current LDS slab
#pragma unroll
    for (int sub = 0; sub < 4; ++sub) {
      FragAB fa, fb;
#pragma unroll
      for (int v = 0; v < 8; ++v) {
        int kb = 32 * sub + ((v < 4) ? 0 : 16) + hi * 8 + (v & 3) * 2;
        fa.u[v] = *(const unsigned int*)&ca[lq * KS + kb];
        fb.u[v] = *(const unsigned int*)&cb[(w * 16 + lq) * BSTR + kb];
      }
      acc = __builtin_amdgcn_wmma_f32_16x16x32_bf16(
          false, fa.v, false, fb.v, (short)0, acc, false, false);
    }

    // commit prefetched slab to the alternate buffer
    if (s + 1 < NSLAB) {
      unsigned short* na = lds_a + ((s + 1) & 1) * (16 * KS);
      unsigned short* nb = lds_b + ((s + 1) & 1) * (64 * BSTR);
#pragma unroll
      for (int it = 0; it < 4; ++it) {
        int idx = tid + 128 * it, r = idx >> 5, k = (idx & 31) * 4;
        *(uint2*)&na[r * KS + k] = ra[it];
      }
#pragma unroll
      for (int it = 0; it < 16; ++it) {
        int idx = tid + 128 * it, n = idx >> 5, k = (idx & 31) * 4;
        *(uint2*)&nb[n * BSTR + k] = rb[it];
      }
    }
    __syncthreads();
  }
  return acc;
}

// ---------------- step kernel 0: decay gates + state prep ----------------
__global__ void __launch_bounds__(128)
k0_prep(const float* __restrict__ inp, const float* __restrict__ x_mean_p,
        const float* __restrict__ b_dg_x, const float* __restrict__ b_dg_h,
        const unsigned short* __restrict__ WdgT,
        const float* __restrict__ hstate, float* __restrict__ xlast,
        float* __restrict__ hdec, unsigned short* __restrict__ Acat, int t)
{
  __shared__ unsigned short lds_a[2 * 16 * KS];
  __shared__ unsigned short lds_b[2 * 64 * BSTR];
  const int m0 = blockIdx.x * 16;
  const int n0 = blockIdx.y * 64;
  // Delta plane: inp[b][2][i][t], stride Tk between features
  const float* dsrc = inp + (size_t)2 * Ik * Tk + t;

  floatx8 acc = gemm_pipe<128>(
      [&](int r, int k) -> uint2 {
        const float* s = dsrc + (size_t)(m0 + r) * 3 * Ik * Tk + (size_t)k * Tk;
        float d0 = s[0], d1 = s[Tk], d2 = s[2 * Tk], d3 = s[3 * Tk];
        uint2 o;
        o.x = (unsigned)f2bf(d0) | ((unsigned)f2bf(d1) << 16);
        o.y = (unsigned)f2bf(d2) | ((unsigned)f2bf(d3) << 16);
        return o;
      },
      WdgT, n0, lds_a, lds_b);

  const int lane = threadIdx.x & 31;
  const int w    = threadIdx.x >> 5;
  const int col  = n0 + w * 16 + (lane & 15);
  const int rb   = m0 + 8 * (lane >> 4);
  const float xm = x_mean_p[0];
#pragma unroll
  for (int j = 0; j < 8; ++j) {
    const int b = rb + j;
    float pre = acc[j] + ((col < Ik) ? b_dg_x[col] : b_dg_h[col - Ik]);
    float g = __expf(-fmaxf(pre, 0.f));
    if (col < Ik) {
      const int i = col;
      size_t base = (size_t)b * 3 * Ik * Tk + (size_t)i * Tk + t;
      float xv = inp[base];
      float mv = inp[base + (size_t)Ik * Tk];
      float xl  = xlast[b * Ik + i];
      float xln = (mv > 0.f) ? xv : xl;
      xlast[b * Ik + i] = xln;
      float xe = mv * xv + (1.f - mv) * (g * xln + (1.f - g) * xm);
      Acat[(size_t)b * ACW + i]       = f2bf(xe);
      Acat[(size_t)b * ACW + 640 + i] = f2bf(mv);
    } else {
      const int nh = col - Ik;
      float hd = g * hstate[b * Hk + nh];
      hdec[b * Hk + nh] = hd;
      Acat[(size_t)b * ACW + 128 + nh] = f2bf(hd);
    }
  }
}

// ---------------- step kernel 1: z and r gates (one 768x1024 GEMM) -------
__global__ void __launch_bounds__(128)
k1_zr(const unsigned short* __restrict__ Acat,
      const unsigned short* __restrict__ WzrT,
      const float* __restrict__ b_z, const float* __restrict__ b_r,
      const float* __restrict__ hdec, float* __restrict__ zbuf,
      unsigned short* __restrict__ AcatW)
{
  __shared__ unsigned short lds_a[2 * 16 * KS];
  __shared__ unsigned short lds_b[2 * 64 * BSTR];
  const int m0 = blockIdx.x * 16;
  const int n0 = blockIdx.y * 64;

  floatx8 acc = gemm_pipe<768>(
      [&](int r, int k) -> uint2 {   // cols 0..767 = [x|hdec|m]
        return *(const uint2*)&Acat[(size_t)(m0 + r) * ACW + k];
      },
      WzrT, n0, lds_a, lds_b);

  const int lane = threadIdx.x & 31;
  const int w    = threadIdx.x >> 5;
  const int col  = n0 + w * 16 + (lane & 15);
  const int rb   = m0 + 8 * (lane >> 4);
#pragma unroll
  for (int j = 0; j < 8; ++j) {
    const int b = rb + j;
    float pre = acc[j] + ((col < Hk) ? b_z[col] : b_r[col - Hk]);
    float s = 1.f / (1.f + __expf(-pre));
    if (col < Hk) {
      zbuf[b * Hk + col] = s;
    } else {
      const int nh = col - Hk;
      AcatW[(size_t)b * ACW + 768 + nh] = f2bf(s * hdec[b * Hk + nh]);
    }
  }
}

// ---------------- step kernel 2: h_tilde + combine + BN partials ---------
__global__ void __launch_bounds__(128)
k2_ht(const unsigned short* __restrict__ Acat,
      const unsigned short* __restrict__ WhT,
      const float* __restrict__ b_h, const float* __restrict__ zbuf,
      const float* __restrict__ hdec, float* __restrict__ hnew,
      float* __restrict__ colsum, float* __restrict__ colsq)
{
  __shared__ unsigned short lds_a[2 * 16 * KS];
  __shared__ unsigned short lds_b[2 * 64 * BSTR];
  const int m0 = blockIdx.x * 16;
  const int n0 = blockIdx.y * 64;

  floatx8 acc = gemm_pipe<768>(
      [&](int r, int k) -> uint2 {
        // A2 = [x(0:128) | rh(768:1280) | m(640:768)]; k is a multiple of 4
        // and region boundaries (128, 640) are too, so groups never straddle.
        int kk = (k >= 128 && k < 640) ? (k + 640) : k;
        return *(const uint2*)&Acat[(size_t)(m0 + r) * ACW + kk];
      },
      WhT, n0, lds_a, lds_b);

  const int lane = threadIdx.x & 31;
  const int w    = threadIdx.x >> 5;
  const int col  = n0 + w * 16 + (lane & 15);
  const int rb   = m0 + 8 * (lane >> 4);
  float ps = 0.f, pq = 0.f;
#pragma unroll
  for (int j = 0; j < 8; ++j) {
    const int b = rb + j;
    float ht = tanhf(acc[j] + b_h[col]);
    float zz = zbuf[b * Hk + col];
    float hd = hdec[b * Hk + col];
    float hn = (1.f - zz) * hd + zz * ht;
    hnew[b * Hk + col] = hn;
    ps += hn;
    pq += hn * hn;
  }
  // fold the two 8-row halves of each column (lane L with lane L^16)
  ps += __shfl_xor(ps, 16, 32);
  pq += __shfl_xor(pq, 16, 32);
  if ((threadIdx.x & 31) < 16) {
    const int rbk = m0 >> 4;                 // row-block id 0..15
    colsum[rbk * Hk + col] = ps;
    colsq [rbk * Hk + col] = pq;
  }
}

// ---------------- step kernel 3: BatchNorm apply + y head ----------------
__global__ void __launch_bounds__(512)
k3_bny(const float* __restrict__ hnew,
       const float* __restrict__ bn_gamma, const float* __restrict__ bn_beta,
       const float* __restrict__ W_hy, const float* __restrict__ b_hy,
       const float* __restrict__ colsum, const float* __restrict__ colsq,
       float* __restrict__ hstate, float* __restrict__ ys,
       float* __restrict__ hs, int t)
{
  __shared__ float red[512];
  const int b = blockIdx.x;
  const int n = threadIdx.x;
  float s = 0.f, q = 0.f;
#pragma unroll
  for (int rb = 0; rb < 16; ++rb) {
    s += colsum[rb * Hk + n];
    q += colsq [rb * Hk + n];
  }
  const float mu  = s * (1.f / Bk);
  const float var = q * (1.f / Bk) - mu * mu;       // biased variance
  const float hn  = hnew[b * Hk + n];
  const float h   = bn_gamma[n] * (hn - mu) * rsqrtf(var + 1e-5f) + bn_beta[n];
  hstate[b * Hk + n] = h;
  hs[(size_t)b * Tk * Hk + (size_t)t * Hk + n] = h;

  for (int o = 0; o < Ok; ++o) {
    red[n] = h * W_hy[n * Ok + o];
    __syncthreads();
    for (int st = 256; st > 0; st >>= 1) {
      if (n < st) red[n] += red[n + st];
      __syncthreads();
    }
    if (n == 0) {
      float v  = red[0] + b_hy[o];
      float sp = fmaxf(v, 0.f) + log1pf(__expf(-fabsf(v)));  // softplus
      ys[(size_t)b * Tk * Ok + t * Ok + o] = sp;
    }
    __syncthreads();
  }
}

// ------- one-time weight packing (f32 -> bf16, concatenated, TRANSPOSED) --
// Destinations are column-major [N][K] so B-slab staging is contiguous.
__global__ void pack_dgT(const float* __restrict__ Wx, const float* __restrict__ Wh_,
                         unsigned short* __restrict__ dst)
{
  for (int idx = blockIdx.x * blockDim.x + threadIdx.x; idx < 640 * 128;
       idx += gridDim.x * blockDim.x) {
    int n = idx / 128, k = idx % 128;
    float v = (n < 128) ? Wx[k * 128 + n] : Wh_[k * 512 + (n - 128)];
    dst[(size_t)n * 128 + k] = f2bf(v);
  }
}

__global__ void pack_zrT(const float* __restrict__ Wxz, const float* __restrict__ Whz,
                         const float* __restrict__ Wmz, const float* __restrict__ Wxr,
                         const float* __restrict__ Whr, const float* __restrict__ Wmr,
                         unsigned short* __restrict__ dst)
{
  for (int idx = blockIdx.x * blockDim.x + threadIdx.x; idx < 1024 * 768;
       idx += gridDim.x * blockDim.x) {
    int n = idx / 768, k = idx % 768;
    bool isz = (n < 512);
    int nn = isz ? n : (n - 512);
    float v;
    if (k < 128)        v = (isz ? Wxz : Wxr)[k * 512 + nn];
    else if (k < 640)   v = (isz ? Whz : Whr)[(k - 128) * 512 + nn];
    else                v = (isz ? Wmz : Wmr)[(k - 640) * 512 + nn];
    dst[(size_t)n * 768 + k] = f2bf(v);
  }
}

__global__ void pack_hT(const float* __restrict__ Wxh, const float* __restrict__ Whh,
                        const float* __restrict__ Wmh, unsigned short* __restrict__ dst)
{
  for (int idx = blockIdx.x * blockDim.x + threadIdx.x; idx < 512 * 768;
       idx += gridDim.x * blockDim.x) {
    int n = idx / 768, k = idx % 768;
    float v;
    if (k < 128)      v = Wxh[k * 512 + n];
    else if (k < 640) v = Whh[(k - 128) * 512 + n];
    else              v = Wmh[(k - 640) * 512 + n];
    dst[(size_t)n * 768 + k] = f2bf(v);
  }
}

__global__ void zero_f32(float* __restrict__ a, int n)
{
  for (int idx = blockIdx.x * blockDim.x + threadIdx.x; idx < n;
       idx += gridDim.x * blockDim.x)
    a[idx] = 0.f;
}

// ---------------- host orchestration ----------------
extern "C" void kernel_launch(void* const* d_in, const int* in_sizes, int n_in,
                              void* d_out, int out_size, void* d_ws, size_t ws_size,
                              hipStream_t stream)
{
  const float* inp     = (const float*)d_in[0];
  const float* x_mean  = (const float*)d_in[1];
  const float* W_dg_x  = (const float*)d_in[2];
  const float* b_dg_x  = (const float*)d_in[3];
  const float* W_dg_h  = (const float*)d_in[4];
  const float* b_dg_h  = (const float*)d_in[5];
  const float* W_xz    = (const float*)d_in[6];
  const float* W_hz    = (const float*)d_in[7];
  const float* W_mz    = (const float*)d_in[8];
  const float* b_z     = (const float*)d_in[9];
  const float* W_xr    = (const float*)d_in[10];
  const float* W_hr    = (const float*)d_in[11];
  const float* W_mr    = (const float*)d_in[12];
  const float* b_r     = (const float*)d_in[13];
  const float* W_xh    = (const float*)d_in[14];
  const float* W_hh    = (const float*)d_in[15];
  const float* W_mh    = (const float*)d_in[16];
  const float* b_h     = (const float*)d_in[17];
  const float* W_hy    = (const float*)d_in[18];
  const float* b_hy    = (const float*)d_in[19];
  const float* bn_g    = (const float*)d_in[20];
  const float* bn_b    = (const float*)d_in[21];

  // workspace carve-out (256B aligned)
  char* p = (char*)d_ws;
  auto carve = [&](size_t bytes) -> void* {
    void* r = (void*)p;
    p += (bytes + 255) & ~(size_t)255;
    return r;
  };
  unsigned short* Acat = (unsigned short*)carve((size_t)Bk * ACW * 2);
  float* hdec          = (float*)carve((size_t)Bk * Hk * 4);
  float* zbuf          = (float*)carve((size_t)Bk * Hk * 4);
  float* hnew          = (float*)carve((size_t)Bk * Hk * 4);
  float* hstate        = (float*)carve((size_t)Bk * Hk * 4);
  float* xlast         = (float*)carve((size_t)Bk * Ik * 4);
  float* colsum        = (float*)carve((size_t)16 * Hk * 4);
  float* colsq         = (float*)carve((size_t)16 * Hk * 4);
  unsigned short* WdgT = (unsigned short*)carve((size_t)640 * 128 * 2);
  unsigned short* WzrT = (unsigned short*)carve((size_t)1024 * 768 * 2);
  unsigned short* WhT  = (unsigned short*)carve((size_t)512 * 768 * 2);

  float* ys = (float*)d_out;
  float* hs = ys + (size_t)Bk * Tk * Ok;

  // one-time packing + state init (re-done every call: deterministic)
  pack_dgT<<<320, 256, 0, stream>>>(W_dg_x, W_dg_h, WdgT);
  pack_zrT<<<1024, 256, 0, stream>>>(W_xz, W_hz, W_mz, W_xr, W_hr, W_mr, WzrT);
  pack_hT<<<1024, 256, 0, stream>>>(W_xh, W_hh, W_mh, WhT);
  zero_f32<<<512, 256, 0, stream>>>(hstate, Bk * Hk);
  zero_f32<<<128, 256, 0, stream>>>(xlast, Bk * Ik);

  for (int t = 0; t < Tk; ++t) {
    k0_prep<<<dim3(Bk / 16, 640 / 64), 128, 0, stream>>>(
        inp, x_mean, b_dg_x, b_dg_h, WdgT, hstate, xlast, hdec, Acat, t);
    k1_zr<<<dim3(Bk / 16, 1024 / 64), 128, 0, stream>>>(
        Acat, WzrT, b_z, b_r, hdec, zbuf, Acat);
    k2_ht<<<dim3(Bk / 16, 512 / 64), 128, 0, stream>>>(
        Acat, WhT, b_h, zbuf, hdec, hnew, colsum, colsq);
    k3_bny<<<Bk, 512, 0, stream>>>(
        hnew, bn_g, bn_b, W_hy, b_hy, colsum, colsq, hstate, ys, hs, t);
  }
}
